// GroupQueryAttention_4157528343097
// MI455X (gfx1250) — matrix operand order repaired and checked by reading, not compiled, not run on hardware
//
#include <hip/hip_runtime.h>
#include <hip/hip_bf16.h>

// ---------------------------------------------------------------------------
// GQA forward for MI455X (gfx1250, wave32, WMMA + async-to-LDS staging).
//   B=2, S=2048, E=2048, Hq=32, Hkv=8, D=64, KV=512
// Pipeline (bf16 WMMA, fp32 accumulation, double-buffered LDS):
//   1) qb = bf16(query @ Wq + bq)          GEMM 4096x2048x2048
//   2) kb = bf16(key   @ Wk + bk)          GEMM 4096x 512x2048
//   3) vb = bf16(value @ Wv + bv)          GEMM 4096x 512x2048
//   4) attn = (q_h @ k_h^T) * 1/8          64 batched GEMMs 2048x2048x64 -> d_out fp32
//   5) row softmax of attn in place        (attn_weights is a required output)
//   6) ho = bf16(attn @ v_h)               64 batched GEMMs 2048x64x2048
//   7) out = ho @ Wo + bo                  GEMM 4096x2048x2048 -> d_out fp32
// Block tile 128x64, BK=32; 8 waves, each wave owns a 32x32 C sub-tile
// (4 v8f accumulators, 4 WMMAs per K-step sharing 2 A-frags and 2 B-frags).
// contiguous bf16 tiles (A, pre-transposed B) are staged with GLOBAL_LOAD_ASYNC_TO_LDS_B128 (ASYNCcnt);
// fp32-source tiles convert in registers and use plain ds stores (DScnt).
// ---------------------------------------------------------------------------

typedef __attribute__((ext_vector_type(8)))  float  v8f;
typedef __attribute__((ext_vector_type(16))) __bf16 v16bf;

#define BM 128
#define BN 64
#define BK 32
#define LDT 40  // LDS row stride in bf16 elements (80B: 16B aligned, staggers banks)

__device__ __forceinline__ unsigned short f2bf(float f) {
  unsigned int u = __float_as_uint(f);
  u += 0x7FFFu + ((u >> 16) & 1u);   // round-to-nearest-even (ignores NaN payload)
  return (unsigned short)(u >> 16);
}

union Frag16 {
  uint4 q[2];
  v16bf v;
};
union Pack8 {
  unsigned short u[8];
  uint4 q;
};

__device__ __forceinline__ Pack8 cvt8(float4 x, float4 y) {
  Pack8 p;
  p.u[0] = f2bf(x.x); p.u[1] = f2bf(x.y); p.u[2] = f2bf(x.z); p.u[3] = f2bf(x.w);
  p.u[4] = f2bf(y.x); p.u[5] = f2bf(y.y); p.u[6] = f2bf(y.z); p.u[7] = f2bf(y.w);
  return p;
}

// Async DMA of 16 contiguous bytes global -> LDS (per lane), tracked by ASYNCcnt.
// VGLOBAL async-load encoding: VDST = LDS byte address VGPR, VADDR = 64-bit
// global address, SADDR = off. LDS byte address = low 32 bits of the generic
// pointer (aperture sits in the high half).
__device__ __forceinline__ void async_copy_b128(const void* gsrc, void* ldst) {
  const unsigned lds = (unsigned)(unsigned long long)ldst;
  asm volatile("global_load_async_to_lds_b128 %0, %1, off"
               :
               : "v"(lds), "v"(gsrc)
               : "memory");
}
__device__ __forceinline__ void async_wait0() {
  asm volatile("s_wait_asynccnt 0" ::: "memory");
}

// C = scale * (A @ B) + bias, batched over blockIdx.z with head-style strides.
//   A: M x K  (row major, fp32 or bf16)
//   B: if B_PRETRANS: stored as (N x K) row major (e.g. K for Q*K^T)
//      else:          stored as (K x N) row major (weights, V)
//   offA = zb*sAb + zh*sAh ; offB = zb*sBb + (zh/group)*sBh ; offC = zb*sCb + zh*sCh
template <bool A_BF16, bool B_BF16, bool B_PRETRANS, bool HAS_BIAS, bool OUT_BF16>
__global__ __launch_bounds__(256) void gqa_gemm_wmma(
    const void* __restrict__ Av, const void* __restrict__ Bv,
    const float* __restrict__ bias, void* __restrict__ Cv,
    int M, int N, int K,
    long long lda, long long ldb, long long ldc,
    long long sAb, long long sAh, long long sBb, long long sBh,
    long long sCb, long long sCh,
    int HperB, int group, float scale) {
  constexpr bool ASYNC_A = A_BF16;                // contiguous bf16 -> async DMA
  constexpr bool ASYNC_B = B_BF16 && B_PRETRANS;  // contiguous bf16 -> async DMA
  constexpr bool ANY_ASYNC = ASYNC_A || ASYNC_B;

  __shared__ __align__(16) unsigned short As[2][BM * LDT];
  __shared__ __align__(16) unsigned short Bs[2][BN * LDT];

  const int z  = blockIdx.z;
  const int zb = z / HperB;
  const int zh = z % HperB;
  const long long offA = (long long)zb * sAb + (long long)zh * sAh;
  const long long offB = (long long)zb * sBb + (long long)(zh / group) * sBh;
  const long long offC = (long long)zb * sCb + (long long)zh * sCh;

  const int nBase = blockIdx.x * BN;
  const int mBase = blockIdx.y * BM;

  const int tid  = threadIdx.x;
  const int lane = tid & 31;
  const int wave = tid >> 5;
  const int l15  = lane & 15;
  const int half = lane >> 4;  // 0: lanes 0-15, 1: lanes 16-31

  // wave tile: 32 rows x 32 cols of the 128x64 block tile (4 waves x 2 waves)
  const int mOff = (wave & 3) * 32;
  const int nOff = (wave >> 2) * 32;

  v8f acc00 = {}, acc01 = {}, acc10 = {}, acc11 = {};

  // staging assignments (256 threads)
  const int aRow  = tid >> 1;        // 0..127, 16 elements each along K
  const int aCol  = (tid & 1) * 16;  // 0 or 16
  const int bRow  = tid >> 2;        // 0..63  (N x K pre-transposed staging)
  const int bCol  = (tid & 3) * 8;   // 0,8,16,24 along K
  const int bkRow = tid >> 3;        // 0..31 along K  (K x N staging)
  const int bnCol = (tid & 7) * 8;   // 0..56 along N

  Pack8 aP0, aP1, bP;  // in-flight tile (registers) for non-async paths

  // Issue global reads for k-tile `kt`; async paths write LDS buffer `b`
  // directly, register paths are published by storeTile(b).
  auto loadTile = [&](int kt, int b) {
    {  // A tile: BM x BK, row major
      const long long base = offA + (long long)(mBase + aRow) * lda + kt + aCol;
      if (A_BF16) {
        const unsigned short* s = (const unsigned short*)Av + base;
        __builtin_prefetch(s + 2 * BK, 0, 1);
        if (ASYNC_A) {
          unsigned short* d = &As[b][aRow * LDT + aCol];
          async_copy_b128(s, d);
          async_copy_b128(s + 8, d + 8);
        } else {
          aP0.q = *(const uint4*)s;
          aP1.q = *(const uint4*)(s + 8);
        }
      } else {
        const float* s = (const float*)Av + base;
        __builtin_prefetch(s + 2 * BK, 0, 1);
        aP0 = cvt8(*(const float4*)s, *(const float4*)(s + 4));
        aP1 = cvt8(*(const float4*)(s + 8), *(const float4*)(s + 12));
      }
    }
    if (B_PRETRANS) {  // source (N x K) row major: rows are output columns
      const long long base = offB + (long long)(nBase + bRow) * ldb + kt + bCol;
      if (B_BF16) {
        const unsigned short* s = (const unsigned short*)Bv + base;
        __builtin_prefetch(s + 2 * BK, 0, 1);
        if (ASYNC_B) {
          async_copy_b128(s, &Bs[b][bRow * LDT + bCol]);
        } else {
          bP.q = *(const uint4*)s;
        }
      } else {
        const float* s = (const float*)Bv + base;
        __builtin_prefetch(s + 2 * BK, 0, 1);
        bP = cvt8(*(const float4*)s, *(const float4*)(s + 4));
      }
    } else {  // source (K x N) row major: transpose while staging
      const long long base = offB + (long long)(kt + bkRow) * ldb + nBase + bnCol;
      if (B_BF16) {
        const unsigned short* s = (const unsigned short*)Bv + base;
        __builtin_prefetch(s + 2 * BK * ldb, 0, 1);
        bP.q = *(const uint4*)s;
      } else {
        const float* s = (const float*)Bv + base;
        __builtin_prefetch(s + 2 * BK * ldb, 0, 1);
        bP = cvt8(*(const float4*)s, *(const float4*)(s + 4));
      }
    }
  };

  auto storeTile = [&](int b) {
    if (!ASYNC_A) {
      *(uint4*)&As[b][aRow * LDT + aCol] = aP0.q;
      *(uint4*)&As[b][aRow * LDT + aCol + 8] = aP1.q;
    }
    if (B_PRETRANS) {
      if (!ASYNC_B) *(uint4*)&Bs[b][bRow * LDT + bCol] = bP.q;
    } else {
#pragma unroll
      for (int j = 0; j < 8; ++j) Bs[b][(bnCol + j) * LDT + bkRow] = bP.u[j];
    }
  };

  const int nK = K / BK;
  loadTile(0, 0);
  storeTile(0);
  if (ANY_ASYNC) async_wait0();
  __syncthreads();

  int buf = 0;
  for (int it = 0; it < nK; ++it) {
    const bool hasNext = (it + 1) < nK;
    if (hasNext) loadTile((it + 1) * BK, buf ^ 1);

    // ---- fragment loads per ISA 16-bit A(16x32) / B(32x16) lane layouts ----
    const unsigned short* Ab = As[buf];
    const unsigned short* Bb = Bs[buf];
    Frag16 a0, a1, b0, b1;
    // A: lane half -> K chunks {half*8 .. +7} and {half*8+16 .. +7}
    const unsigned short* ap0 = &Ab[(mOff + l15) * LDT + half * 8];
    a0.q[0] = *(const uint4*)ap0;
    a0.q[1] = *(const uint4*)(ap0 + 16);
    const unsigned short* ap1 = ap0 + 16 * LDT;
    a1.q[0] = *(const uint4*)ap1;
    a1.q[1] = *(const uint4*)(ap1 + 16);
    // B: lane half -> K chunks {half*8 .. +7} and {half*8+16 .. +7}
    const unsigned short* bp0 = &Bb[(nOff + l15) * LDT + half * 8];
    b0.q[0] = *(const uint4*)bp0;
    b0.q[1] = *(const uint4*)(bp0 + 16);
    const unsigned short* bp1 = bp0 + 16 * LDT;
    b1.q[0] = *(const uint4*)bp1;
    b1.q[1] = *(const uint4*)(bp1 + 16);

    acc00 = __builtin_amdgcn_wmma_f32_16x16x32_bf16(false, a0.v, false, b0.v,
                                                    (short)0, acc00, false, false);
    acc01 = __builtin_amdgcn_wmma_f32_16x16x32_bf16(false, a0.v, false, b1.v,
                                                    (short)0, acc01, false, false);
    acc10 = __builtin_amdgcn_wmma_f32_16x16x32_bf16(false, a1.v, false, b0.v,
                                                    (short)0, acc10, false, false);
    acc11 = __builtin_amdgcn_wmma_f32_16x16x32_bf16(false, a1.v, false, b1.v,
                                                    (short)0, acc11, false, false);

    if (hasNext) storeTile(buf ^ 1);
    if (ANY_ASYNC) async_wait0();  // fence this wave's DMAs into buf^1
    __syncthreads();               // publish buf^1 to the workgroup
    buf ^= 1;
  }

  // ---- epilogue: C/D layout VGPR r -> row r (lanes 0-15) / row r+8 (16-31) ----
  const int col0 = nBase + nOff + l15;
  const int col1 = col0 + 16;
  float bias0 = 0.f, bias1 = 0.f;
  if (HAS_BIAS) {
    bias0 = bias[col0];
    bias1 = bias[col1];
  }
#pragma unroll
  for (int r = 0; r < 8; ++r) {
    const int rr = half ? (r + 8) : r;
    const long long row0 = (long long)(mBase + mOff + rr);
    const long long row1 = row0 + 16;
    float v00 = acc00[r] * scale + bias0;
    float v01 = acc01[r] * scale + bias1;
    float v10 = acc10[r] * scale + bias0;
    float v11 = acc11[r] * scale + bias1;
    const long long i00 = offC + row0 * ldc + col0;
    const long long i01 = offC + row0 * ldc + col1;
    const long long i10 = offC + row1 * ldc + col0;
    const long long i11 = offC + row1 * ldc + col1;
    if (OUT_BF16) {
      unsigned short* C = (unsigned short*)Cv;
      C[i00] = f2bf(v00); C[i01] = f2bf(v01);
      C[i10] = f2bf(v10); C[i11] = f2bf(v11);
    } else {
      float* C = (float*)Cv;
      C[i00] = v00; C[i01] = v01;
      C[i10] = v10; C[i11] = v11;
    }
  }
}

// Row-wise softmax, one 256-thread block per row of 2048 floats (in place).
__global__ __launch_bounds__(256) void gqa_softmax_rows(float* __restrict__ p) {
  const long long row = blockIdx.x;
  float* x = p + row * 2048ll;
  const int tid = threadIdx.x;
  __shared__ float red[256];

  float loc[8];
  float m = -3.402823e38f;
#pragma unroll
  for (int i = 0; i < 8; ++i) {
    loc[i] = x[tid + i * 256];
    m = fmaxf(m, loc[i]);
  }
  red[tid] = m;
  __syncthreads();
  for (int s = 128; s > 0; s >>= 1) {
    if (tid < s) red[tid] = fmaxf(red[tid], red[tid + s]);
    __syncthreads();
  }
  m = red[0];
  __syncthreads();

  float sum = 0.f;
#pragma unroll
  for (int i = 0; i < 8; ++i) {
    loc[i] = __expf(loc[i] - m);
    sum += loc[i];
  }
  red[tid] = sum;
  __syncthreads();
  for (int s = 128; s > 0; s >>= 1) {
    if (tid < s) red[tid] += red[tid + s];
    __syncthreads();
  }
  const float inv = 1.0f / red[0];
#pragma unroll
  for (int i = 0; i < 8; ++i) x[tid + i * 256] = loc[i] * inv;
}

extern "C" void kernel_launch(void* const* d_in, const int* in_sizes, int n_in,
                              void* d_out, int out_size, void* d_ws, size_t ws_size,
                              hipStream_t stream) {
  (void)in_sizes; (void)n_in; (void)out_size; (void)ws_size;

  constexpr int Bb = 2, S = 2048, E = 2048, H = 32, G = 4, KV = 512, D = 64;
  constexpr int BS = Bb * S;  // 4096 rows of activations

  const float* query = (const float*)d_in[0];
  const float* key_  = (const float*)d_in[1];
  const float* value = (const float*)d_in[2];
  const float* Wq = (const float*)d_in[3];
  const float* bq = (const float*)d_in[4];
  const float* Wk = (const float*)d_in[5];
  const float* bk = (const float*)d_in[6];
  const float* Wv = (const float*)d_in[7];
  const float* bv = (const float*)d_in[8];
  const float* Wo = (const float*)d_in[9];
  const float* bo = (const float*)d_in[10];

  float* out  = (float*)d_out;                       // (B,S,E)
  float* attn = out + (long long)Bb * S * E;         // (B,H,S,S)

  // workspace (bf16 intermediates): 40 MB total
  unsigned short* qb = (unsigned short*)d_ws;        // (BS, E)
  unsigned short* kb = qb + (size_t)BS * E;          // (BS, KV)
  unsigned short* vb = kb + (size_t)BS * KV;         // (BS, KV)
  unsigned short* ho = vb + (size_t)BS * KV;         // (BS, E) head-major concat

  const dim3 blk(256);

  // 1-3) input projections (fp32 A, fp32 weights K x N, bias, bf16 out)
  gqa_gemm_wmma<false, false, false, true, true>
      <<<dim3(E / BN, BS / BM, 1), blk, 0, stream>>>(
          query, Wq, bq, qb, BS, E, E, E, E, E, 0, 0, 0, 0, 0, 0, 1, 1, 1.0f);
  gqa_gemm_wmma<false, false, false, true, true>
      <<<dim3(KV / BN, BS / BM, 1), blk, 0, stream>>>(
          key_, Wk, bk, kb, BS, KV, E, E, KV, KV, 0, 0, 0, 0, 0, 0, 1, 1, 1.0f);
  gqa_gemm_wmma<false, false, false, true, true>
      <<<dim3(KV / BN, BS / BM, 1), blk, 0, stream>>>(
          value, Wv, bv, vb, BS, KV, E, E, KV, KV, 0, 0, 0, 0, 0, 0, 1, 1, 1.0f);

  // 4) scores = q_h @ k_h^T * 1/sqrt(D); K-matrix rows ARE B^T (pre-transposed)
  //    -> both A and B tiles staged with async-to-LDS DMA
  gqa_gemm_wmma<true, true, true, false, false>
      <<<dim3(S / BN, S / BM, Bb * H), blk, 0, stream>>>(
          qb, kb, nullptr, attn, S, S, D,
          /*lda*/ E, /*ldb*/ KV, /*ldc*/ S,
          /*sAb*/ (long long)S * E, /*sAh*/ D,
          /*sBb*/ (long long)S * KV, /*sBh*/ D,
          /*sCb*/ (long long)H * S * S, /*sCh*/ (long long)S * S,
          H, G, 0.125f);

  // 5) softmax rows of attn in place (attn_weights output)
  gqa_softmax_rows<<<dim3((unsigned)(Bb * H * S)), blk, 0, stream>>>(attn);

  // 6) head_out = attn @ v_h (fp32 A, bf16 V as K x N, bf16 out)
  gqa_gemm_wmma<false, true, false, false, true>
      <<<dim3(D / BN, S / BM, Bb * H), blk, 0, stream>>>(
          attn, vb, nullptr, ho, S, D, S,
          /*lda*/ S, /*ldb*/ KV, /*ldc*/ E,
          /*sAb*/ (long long)H * S * S, /*sAh*/ (long long)S * S,
          /*sBb*/ (long long)S * KV, /*sBh*/ D,
          /*sCb*/ (long long)S * E, /*sCh*/ D,
          H, G, 1.0f);

  // 7) out = ho @ Wo + bo (bf16 A async, fp32 weights, fp32 out)
  gqa_gemm_wmma<true, false, false, true, false>
      <<<dim3(E / BN, BS / BM, 1), blk, 0, stream>>>(
          ho, Wo, bo, out, BS, E, E, E, E, E, 0, 0, 0, 0, 0, 0, 1, 1, 1.0f);
}
